// GAT_48584670052992
// MI455X (gfx1250) — matrix operand order repaired and checked by reading, not compiled, not run on hardware
//
#include <hip/hip_runtime.h>
#include <hip/hip_bf16.h>

typedef __attribute__((ext_vector_type(16))) _Float16 v16h;
typedef __attribute__((ext_vector_type(8)))  float    v8f;

#define LN_EPS    1e-5f
#define NEG_SLOPE 0.2f

// ---- helpers -------------------------------------------------------------

// relaxed agent-scope f32 atomic add -> global_atomic_add_f32 (no CAS loop)
static __device__ __forceinline__ void atomAddF(float* p, float v) {
  __hip_atomic_fetch_add(p, v, __ATOMIC_RELAXED, __HIP_MEMORY_SCOPE_AGENT);
}

// order-preserving float <-> u32 map so segment-max can use GLOBAL_ATOMIC_MAX_U32,
// with memset(0) as the -inf identity.
static __device__ __forceinline__ unsigned fenc(float x) {
  unsigned u = __float_as_uint(x);
  return (u & 0x80000000u) ? ~u : (u | 0x80000000u);
}
static __device__ __forceinline__ float fdec(unsigned u) {
  return __uint_as_float((u & 0x80000000u) ? (u & 0x7FFFFFFFu) : ~u);
}

static __device__ __forceinline__ float lrelu(float x) {
  return x >= 0.f ? x : NEG_SLOPE * x;
}

// ---- weight prep: f32 -> f16, transposed so B fragments are contiguous ----

__global__ void prep_weights(const float* __restrict__ W0, const float* __restrict__ W1,
                             _Float16* __restrict__ Wt0, _Float16* __restrict__ Wt1) {
  int t = blockIdx.x * blockDim.x + threadIdx.x;
  if (t < 128 * 128) { int c = t >> 7, k = t & 127; Wt0[c * 128 + k] = (_Float16)W0[k * 128 + c]; }
  if (t < 64 * 128)  { int c = t >> 7, k = t & 127; Wt1[c * 128 + k] = (_Float16)W1[k * 64 + c]; }
}

// ---- WMMA GEMM: C[M,Ncols] = A[M,Kd] (f32, cvt to f16) x Bt (f16, [Ncols][Kd]) ----
// One wave per 16x16 output tile; blockDim.x = (Ncols/16)*32.
__global__ void gemm_wmma_f16(const float* __restrict__ A, const _Float16* __restrict__ Bt,
                              float* __restrict__ C, int M, int Kd, int Ncols) {
  const int lane = threadIdx.x & 31;
  const int wv   = threadIdx.x >> 5;
  const int row0 = blockIdx.x * 16;
  const int col0 = wv * 16;
  const int hi   = lane >> 4;    // half-wave select
  const int lid  = lane & 15;

  int r = row0 + lid;
  if (r >= M) r = M - 1;         // clamp reads; stores predicated below

  v8f acc = {};
  for (int k0 = 0; k0 < Kd; k0 += 32) {
    // A fragment: lanes 0-15 hold K {k0..k0+15},
    // lanes 16-31 hold K {k0+16..k0+31} (matches B's contiguous order).
    const float* ap = A + (size_t)r * Kd + k0 + hi * 16;
    v16h a;
#pragma unroll
    for (int j = 0; j < 8; ++j) {
      a[j]     = (_Float16)ap[j];
      a[j + 8] = (_Float16)ap[j + 8];
    }
    // B fragment: column (col0+lid), 16 contiguous K values per half-wave.
    v16h b = *(const v16h*)(Bt + (size_t)(col0 + lid) * Kd + k0 + hi * 16);
    acc = __builtin_amdgcn_wmma_f32_16x16x32_f16(false, a, false, b, (short)0, acc,
                                                 false, false);
  }
  // C layout: VGPR j -> M = row0 + j + 8*hi, N = col0 + lid.
  const int c  = col0 + lid;
  const int mb = row0 + hi * 8;
#pragma unroll
  for (int j = 0; j < 8; ++j) {
    int m = mb + j;
    if (m < M) C[(size_t)m * Ncols + c] = acc[j];
  }
}

// ---- el/er projections: el[n,h] = dot(feat[n,h,:], al[h,:]) ----------------

__global__ void eler_kernel(const float* __restrict__ feat, const float* __restrict__ al,
                            const float* __restrict__ ar, float* __restrict__ el,
                            float* __restrict__ er, int N, int Hh) {
  int n = blockIdx.x * blockDim.x + threadIdx.x;
  if (n >= N) return;
  for (int h = 0; h < Hh; ++h) {
    const float* f = feat + (size_t)n * Hh * 64 + h * 64;
    float sl = 0.f, sr = 0.f;
#pragma unroll 4
    for (int d = 0; d < 64; ++d) { sl += f[d] * al[h * 64 + d]; sr += f[d] * ar[h * 64 + d]; }
    el[n * Hh + h] = sl;
    er[n * Hh + h] = sr;
  }
}

// ---- edge pass 1: segment max via encoded u32 atomic max -------------------

__global__ void edge_max(const int* __restrict__ src, const int* __restrict__ dst,
                         const float* __restrict__ el, const float* __restrict__ er,
                         unsigned* __restrict__ m, int E, int Hh) {
  int t = blockIdx.x * blockDim.x + threadIdx.x;
  if (t >= E * Hh) return;
  int e = t / Hh, h = t - e * Hh;
  int s = src[e], d = dst[e];
  float x = lrelu(el[s * Hh + h] + er[d * Hh + h]);
  atomicMax(m + d * Hh + h, fenc(x));
}

// ---- edge pass 2 (layer 0, H=2, 128 dims): wave per edge, 4 dims/lane ------

__global__ void edge_acc_l0(const int* __restrict__ src, const int* __restrict__ dst,
                            const float* __restrict__ el, const float* __restrict__ er,
                            const unsigned* __restrict__ m, const float* __restrict__ feat,
                            float* __restrict__ rst, float* __restrict__ den, int E) {
  int g = blockIdx.x * blockDim.x + threadIdx.x;
  int e = g >> 5, lane = g & 31;
  if (e >= E) return;
  int s = src[e], d = dst[e];
  int h = lane >> 4;  // dims [lane*4, lane*4+3]; head boundary at dim 64 == lane 16
  float x = lrelu(el[s * 2 + h] + er[d * 2 + h]);
  float p = __expf(x - fdec(m[d * 2 + h]));
  if (lane == 0 || lane == 16) atomAddF(den + d * 2 + h, p);
  const float4 f = *(const float4*)(feat + (size_t)s * 128 + lane * 4);
  float* rp = rst + (size_t)d * 128 + lane * 4;
  atomAddF(rp + 0, p * f.x);
  atomAddF(rp + 1, p * f.y);
  atomAddF(rp + 2, p * f.z);
  atomAddF(rp + 3, p * f.w);
}

// ---- edge pass 2 (layer 1, H=1, 64 dims): wave per edge, 2 dims/lane -------

__global__ void edge_acc_l1(const int* __restrict__ src, const int* __restrict__ dst,
                            const float* __restrict__ el, const float* __restrict__ er,
                            const unsigned* __restrict__ m, const float* __restrict__ feat,
                            float* __restrict__ rst, float* __restrict__ den, int E) {
  int g = blockIdx.x * blockDim.x + threadIdx.x;
  int e = g >> 5, lane = g & 31;
  if (e >= E) return;
  int s = src[e], d = dst[e];
  float x = lrelu(el[s] + er[d]);
  float p = __expf(x - fdec(m[d]));
  if (lane == 0) atomAddF(den + d, p);
  const float2 f = *(const float2*)(feat + (size_t)s * 64 + lane * 2);
  float* rp = rst + (size_t)d * 64 + lane * 2;
  atomAddF(rp + 0, p * f.x);
  atomAddF(rp + 1, p * f.y);
}

// ---- node pass: normalize by denom, +bias, (relu), LayerNorm ---------------
// DIM threads per block (one node per block), wave32 shuffle reductions.
template <int DIM, int HEADS, bool RELU>
__global__ void node_norm(const float* __restrict__ rst, const float* __restrict__ den,
                          const float* __restrict__ bias, const float* __restrict__ g,
                          const float* __restrict__ be, float* __restrict__ out, int N) {
  __shared__ float sh[DIM / 32];
  __shared__ float shb;
  const int n = blockIdx.x;
  const int d = threadIdx.x;
  const int lane = d & 31, wv = d >> 5;
  constexpr int NW = DIM / 32;

  float dn = den[n * HEADS + d / (DIM / HEADS)];
  float v  = rst[(size_t)n * DIM + d];
  v = dn > 0.f ? v / dn : 0.f;
  v += bias[d];
  if (RELU) v = v > 0.f ? v : 0.f;

  // mean
  float s = v;
  for (int off = 16; off; off >>= 1) s += __shfl_down(s, off, 32);
  if (lane == 0) sh[wv] = s;
  __syncthreads();
  if (d == 0) { float t = 0.f; for (int i = 0; i < NW; ++i) t += sh[i]; shb = t; }
  __syncthreads();
  float mu = shb * (1.f / DIM);
  __syncthreads();

  // variance
  float c = v - mu;
  s = c * c;
  for (int off = 16; off; off >>= 1) s += __shfl_down(s, off, 32);
  if (lane == 0) sh[wv] = s;
  __syncthreads();
  if (d == 0) { float t = 0.f; for (int i = 0; i < NW; ++i) t += sh[i]; shb = t; }
  __syncthreads();
  float var = shb * (1.f / DIM);

  out[(size_t)n * DIM + d] = c * rsqrtf(var + LN_EPS) * g[d] + be[d];
}

// ---- launcher --------------------------------------------------------------

extern "C" void kernel_launch(void* const* d_in, const int* in_sizes, int n_in,
                              void* d_out, int out_size, void* d_ws, size_t ws_size,
                              hipStream_t stream) {
  const float* h   = (const float*)d_in[0];
  const float* W0  = (const float*)d_in[1];
  const float* al0 = (const float*)d_in[2];
  const float* ar0 = (const float*)d_in[3];
  const float* b0  = (const float*)d_in[4];
  const float* W1  = (const float*)d_in[5];
  const float* al1 = (const float*)d_in[6];
  const float* ar1 = (const float*)d_in[7];
  const float* b1  = (const float*)d_in[8];
  const float* g0  = (const float*)d_in[9];
  const float* be0 = (const float*)d_in[10];
  const float* g1  = (const float*)d_in[11];
  const float* be1 = (const float*)d_in[12];
  const int*   src = (const int*)d_in[13];
  const int*   dst = (const int*)d_in[14];
  const int N = in_sizes[0] / 128;   // 50000
  const int E = in_sizes[13];        // 800000
  float* out = (float*)d_out;
  (void)n_in; (void)out_size; (void)ws_size;

  // workspace carve-out (256B aligned)
  char* ws = (char*)d_ws;
  size_t off = 0;
  auto carve = [&](size_t bytes) -> char* {
    char* p = ws + off;
    off = (off + bytes + 255) & ~(size_t)255;
    return p;
  };
  float*    feat0 = (float*)carve((size_t)N * 128 * 4);  // layer-0 feats, then x
  float*    buf2  = (float*)carve((size_t)N * 128 * 4);  // rst0; later feat1|rst1
  float*    el0   = (float*)carve((size_t)N * 2 * 4);
  float*    er0   = (float*)carve((size_t)N * 2 * 4);
  unsigned* m0    = (unsigned*)carve((size_t)N * 2 * 4);
  float*    den0  = (float*)carve((size_t)N * 2 * 4);
  float*    el1   = (float*)carve((size_t)N * 4);
  float*    er1   = (float*)carve((size_t)N * 4);
  unsigned* m1    = (unsigned*)carve((size_t)N * 4);
  float*    den1  = (float*)carve((size_t)N * 4);
  _Float16* Wt0   = (_Float16*)carve(128 * 128 * 2);
  _Float16* Wt1   = (_Float16*)carve(64 * 128 * 2);

  const int mt = (N + 15) / 16;

  prep_weights<<<(128 * 128 + 255) / 256, 256, 0, stream>>>(W0, W1, Wt0, Wt1);

  // ---- layer 0 ----
  hipMemsetAsync(m0, 0, (size_t)N * 2 * 4, stream);          // fenc identity (-inf)
  hipMemsetAsync(den0, 0, (size_t)N * 2 * 4, stream);
  hipMemsetAsync(buf2, 0, (size_t)N * 128 * 4, stream);      // rst0 = 0
  gemm_wmma_f16<<<mt, 256, 0, stream>>>(h, Wt0, feat0, N, 128, 128);
  eler_kernel<<<(N + 255) / 256, 256, 0, stream>>>(feat0, al0, ar0, el0, er0, N, 2);
  edge_max<<<(E * 2 + 255) / 256, 256, 0, stream>>>(src, dst, el0, er0, m0, E, 2);
  edge_acc_l0<<<(int)(((size_t)E * 32 + 255) / 256), 256, 0, stream>>>(
      src, dst, el0, er0, m0, feat0, buf2, den0, E);
  node_norm<128, 2, true><<<N, 128, 0, stream>>>(buf2, den0, b0, g0, be0, feat0, N);

  // ---- layer 1 ----
  float* feat1 = buf2;
  float* rst1  = buf2 + (size_t)N * 64;
  hipMemsetAsync(m1, 0, (size_t)N * 4, stream);
  hipMemsetAsync(den1, 0, (size_t)N * 4, stream);
  hipMemsetAsync(rst1, 0, (size_t)N * 64 * 4, stream);
  gemm_wmma_f16<<<mt, 128, 0, stream>>>(feat0, Wt1, feat1, N, 128, 64);
  eler_kernel<<<(N + 255) / 256, 256, 0, stream>>>(feat1, al1, ar1, el1, er1, N, 1);
  edge_max<<<(E + 255) / 256, 256, 0, stream>>>(src, dst, el1, er1, m1, E, 1);
  edge_acc_l1<<<(int)(((size_t)E * 32 + 255) / 256), 256, 0, stream>>>(
      src, dst, el1, er1, m1, feat1, rst1, den1, E);
  node_norm<64, 1, false><<<N, 64, 0, stream>>>(rst1, den1, b1, g1, be1, out, N);
}
